// LocalWriteLayer_51256139710509
// MI455X (gfx1250) — matrix operand order repaired and checked by reading, not compiled, not run on hardware
//
#include <hip/hip_runtime.h>
#include <hip/hip_bf16.h>

#define C_CH  256
#define HEADS 8
#define HW    65536          // 256*256
#define WPIX  256
#define PIXT  64             // pixels per block tile
#define LDSW  264            // padded row stride (bf16 elems): 528B, 16B aligned, bank step 4

typedef __attribute__((ext_vector_type(16))) __bf16 v16bf;
typedef __attribute__((ext_vector_type(8)))  float  v8f;
typedef __attribute__((ext_vector_type(4)))  int    i4v;

union FragU { i4v q[2]; v16bf v; };

__device__ __forceinline__ unsigned short f2bf(float f) {
  unsigned int u = __float_as_uint(f);
  unsigned int r = u + 0x7fffu + ((u >> 16) & 1u);   // round-to-nearest-even
  return (unsigned short)(r >> 16);
}
__device__ __forceinline__ float bf2f(unsigned int hs) {
  return __uint_as_float(hs << 16);
}

// ---------------------------------------------------------------------------
// Setup: fold BN into GEMM weights, convert to bf16, transpose w_values.
// which: 0=w_proj(bn1) 1=w_mlp1(bn2) 2=w_mlp2 3=w_values(transposed)
// ---------------------------------------------------------------------------
__global__ __launch_bounds__(256) void fold_kernel(
    const float* __restrict__ w_proj, const float* __restrict__ b_proj,
    const float* __restrict__ w_mlp1, const float* __restrict__ b_mlp1,
    const float* __restrict__ w_mlp2, const float* __restrict__ b_mlp2,
    const float* __restrict__ w_values, const float* __restrict__ b_values,
    const float* __restrict__ g1, const float* __restrict__ bb1,
    const float* __restrict__ m1, const float* __restrict__ v1,
    const float* __restrict__ g2, const float* __restrict__ bb2,
    const float* __restrict__ m2, const float* __restrict__ v2,
    unsigned short* __restrict__ wp, unsigned short* __restrict__ wm1,
    unsigned short* __restrict__ wm2, unsigned short* __restrict__ wvT,
    float* __restrict__ biasp, float* __restrict__ biasm1,
    float* __restrict__ biasm2, float* __restrict__ biasv)
{
  int o = blockIdx.x, c = threadIdx.x, which = blockIdx.y;
  __shared__ float red[256];
  float beta = 0.f, b0 = 0.f, wv = 0.f;
  if (which == 0) {
    float inv = g1[c] * rsqrtf(v1[c] + 1e-5f);
    beta = bb1[c] - m1[c] * inv;
    wv = w_proj[o*256 + c]; b0 = b_proj[o];
    wp[o*256 + c] = f2bf(wv * inv);
  } else if (which == 1) {
    float inv = g2[c] * rsqrtf(v2[c] + 1e-5f);
    beta = bb2[c] - m2[c] * inv;
    wv = w_mlp1[o*256 + c]; b0 = b_mlp1[o];
    wm1[o*256 + c] = f2bf(wv * inv);
  } else if (which == 2) {
    wv = w_mlp2[o*256 + c]; b0 = b_mlp2[o];
    wm2[o*256 + c] = f2bf(wv);
  } else {
    b0 = b_values[o];
    wvT[c*256 + o] = f2bf(w_values[o*256 + c]);   // transpose for coalesced reads
    wv = 0.f;
  }
  red[c] = wv * beta;
  __syncthreads();
  for (int s = 128; s > 0; s >>= 1) { if (c < s) red[c] += red[c + s]; __syncthreads(); }
  if (c == 0) {
    float bias = b0 + red[0];
    if (which == 0) biasp[o] = bias;
    else if (which == 1) biasm1[o] = bias;
    else if (which == 2) biasm2[o] = bias;
    else biasv[o] = bias;
  }
}

// ---------------------------------------------------------------------------
// vals[b,n,c] = xlr[b,n,:] . w_values[c,:] + b_values[c]   (tiny GEMM)
// ---------------------------------------------------------------------------
__global__ __launch_bounds__(256) void vals_kernel(
    const float* __restrict__ xlr, const unsigned short* __restrict__ wvT,
    const float* __restrict__ biasv, float* __restrict__ vals)
{
  int bn = blockIdx.x, o = threadIdx.x;
  __shared__ float xl[256];
  xl[o] = xlr[bn*256 + o];
  __syncthreads();
  float a = biasv[o];
  for (int k = 0; k < 256; ++k)
    a += xl[k] * bf2f(wvT[k*256 + o]);     // lanes read consecutive o: coalesced
  vals[bn*256 + o] = a;
}

// ---------------------------------------------------------------------------
// One 256x64 GEMM: acc[2][4] (+=) W[256,256]_bf16 x buf[pix][K]_bf16
// Wave owns M-tiles {mtb, mtb+16}, N-tiles 0..3.  m=lane&15, h=lane>>4.
// ---------------------------------------------------------------------------
__device__ __forceinline__ void gemm_256x64(
    const unsigned short* __restrict__ W, const unsigned short* buf,
    int mtb, int m, int h, v8f acc[2][4])
{
#pragma unroll
  for (int kc = 0; kc < 8; ++kc) {
    int kb = kc * 32;
    FragU a0, a1;
    // A frag: row = tile_base + m ; K runs [kb+h*8, +8) and [kb+16+h*8, +8)
    const unsigned short* wr0 = W + (mtb + m) * 256 + kb + h * 8;
    const unsigned short* wr1 = wr0 + 16 * 256;
    a0.q[0] = *(const i4v*)wr0;  a0.q[1] = *(const i4v*)(wr0 + 16);
    a1.q[0] = *(const i4v*)wr1;  a1.q[1] = *(const i4v*)(wr1 + 16);
#pragma unroll
    for (int ni = 0; ni < 4; ++ni) {
      FragU bfr;  // B frag: col = ni*16+m ; K runs [kb+h*8, +8) and [kb+16+h*8, +8)
      const unsigned short* br = buf + (ni * 16 + m) * LDSW + kb + h * 8;
      bfr.q[0] = *(const i4v*)br;
      bfr.q[1] = *(const i4v*)(br + 16);
      acc[0][ni] = __builtin_amdgcn_wmma_f32_16x16x32_bf16(
          false, a0.v, false, bfr.v, (short)0, acc[0][ni], false, false);
      acc[1][ni] = __builtin_amdgcn_wmma_f32_16x16x32_bf16(
          false, a1.v, false, bfr.v, (short)0, acc[1][ni], false, false);
    }
  }
}

// ---------------------------------------------------------------------------
// Fused main kernel: one block = 64 consecutive pixels of one image row.
// ---------------------------------------------------------------------------
__global__ __launch_bounds__(256) void fused_kernel(
    const float* __restrict__ xhr, const float* __restrict__ w_logits,
    const float* __restrict__ b_logits,
    const unsigned short* __restrict__ wp, const unsigned short* __restrict__ wm1,
    const unsigned short* __restrict__ wm2,
    const float* __restrict__ biasp, const float* __restrict__ biasm1,
    const float* __restrict__ biasm2,
    const float* __restrict__ vals, float* __restrict__ out)
{
  __shared__ __align__(16) unsigned short bufA[PIXT * LDSW];  // [pix][ch] bf16
  __shared__ __align__(16) unsigned short bufB[PIXT * LDSW];
  __shared__ float wl[HEADS * 256];
  __shared__ float sig[HEADS * PIXT];
  __shared__ float partial[4 * HEADS * PIXT];
  __shared__ float valsl[4 * 256];

  int t = threadIdx.x;
  int l = t & 31, wvid = t >> 5;
  int bid = blockIdx.x;
  int b   = bid >> 10;
  int rr  = bid & 1023;
  int row = rr >> 2;
  int x0  = (rr & 3) << 6;
  const float* xbase = xhr + b * (C_CH * HW) + row * WPIX + x0;
  float*       obase = out + b * (C_CH * HW) + row * WPIX + x0;

#pragma unroll
  for (int i = 0; i < 8; ++i) wl[i * 256 + t] = w_logits[i * 256 + t];

  int n0 = (row >> 4) * 16 + (x0 >> 4);           // first window index
#pragma unroll
  for (int i = 0; i < 4; ++i) {
    int idx = t + i * 256;
    valsl[idx] = vals[(b * 256 + n0 + (idx >> 8)) * 256 + (idx & 255)];
  }

  int p  = t & 63;
  int cg = t >> 6;
  for (int i = 0; i < 64; ++i) {                   // x tile -> bufA (transposed)
    int c = i * 4 + cg;
    bufA[p * LDSW + c] = f2bf(xbase[c * HW + p]);
  }
  __syncthreads();

  { // logits partials: group cg handles channels [cg*64, cg*64+64)
    float acch[8];
#pragma unroll
    for (int hh = 0; hh < 8; ++hh) acch[hh] = 0.f;
    for (int i = 0; i < 32; ++i) {
      unsigned int u = *(const unsigned int*)&bufA[p * LDSW + cg * 64 + i * 2];
      float xa = bf2f(u & 0xffffu), xb = bf2f(u >> 16);
      int c0 = cg * 64 + i * 2;
#pragma unroll
      for (int hh = 0; hh < 8; ++hh)
        acch[hh] += wl[hh * 256 + c0] * xa + wl[hh * 256 + c0 + 1] * xb;
    }
#pragma unroll
    for (int hh = 0; hh < 8; ++hh) partial[(cg * 8 + hh) * 64 + p] = acch[hh];
  }
  __syncthreads();

#pragma unroll
  for (int j = 0; j < 2; ++j) {                    // reduce + sigmoid
    int idx = t + j * 256;
    int hh = idx >> 6, pp = idx & 63;
    float s = partial[(0 * 8 + hh) * 64 + pp] + partial[(1 * 8 + hh) * 64 + pp]
            + partial[(2 * 8 + hh) * 64 + pp] + partial[(3 * 8 + hh) * 64 + pp]
            + b_logits[hh];
    sig[hh * 64 + pp] = 1.f / (1.f + __expf(-s));
  }
  __syncthreads();

  for (int i = 0; i < 64; ++i) {                   // read = sig * vals -> bufB
    int c = cg * 64 + i;
    float r = sig[(c >> 5) * 64 + p] * valsl[(p >> 4) * 256 + c];
    bufB[p * LDSW + c] = f2bf(r);
  }
  __syncthreads();

  int m = l & 15, h = l >> 4;
  int mtb = wvid * 32;
  v8f acc[2][4], resid[2][4];

  // -------- stage 1: xhr2 = xhr + Wp' @ read + biasp --------
#pragma unroll
  for (int mi = 0; mi < 2; ++mi) {
    int chb = mtb + mi * 16 + h * 8;
#pragma unroll
    for (int i = 0; i < 8; ++i) {
      float bv = biasp[chb + i];
#pragma unroll
      for (int ni = 0; ni < 4; ++ni) acc[mi][ni][i] = bv;
    }
  }
  gemm_256x64(wp, bufB, mtb, m, h, acc);
#pragma unroll
  for (int mi = 0; mi < 2; ++mi)
#pragma unroll
    for (int ni = 0; ni < 4; ++ni)
#pragma unroll
      for (int i = 0; i < 8; ++i) {
        int ch  = mtb + mi * 16 + i + h * 8;
        int pix = ni * 16 + m;
        float v2 = acc[mi][ni][i] + xbase[ch * HW + pix];
        resid[mi][ni][i] = v2;                     // keep xhr2 in registers
        bufA[pix * LDSW + ch] = f2bf(v2);
      }
  __syncthreads();

  // -------- stage 2: hmid = gelu(Wm1' @ xhr2 + biasm1) --------
#pragma unroll
  for (int mi = 0; mi < 2; ++mi) {
    int chb = mtb + mi * 16 + h * 8;
#pragma unroll
    for (int i = 0; i < 8; ++i) {
      float bv = biasm1[chb + i];
#pragma unroll
      for (int ni = 0; ni < 4; ++ni) acc[mi][ni][i] = bv;
    }
  }
  gemm_256x64(wm1, bufA, mtb, m, h, acc);
#pragma unroll
  for (int mi = 0; mi < 2; ++mi)
#pragma unroll
    for (int ni = 0; ni < 4; ++ni)
#pragma unroll
      for (int i = 0; i < 8; ++i) {
        int ch  = mtb + mi * 16 + i + h * 8;
        int pix = ni * 16 + m;
        float g = acc[mi][ni][i];
        g = 0.5f * g * (1.f + erff(g * 0.70710678f));   // exact GELU
        bufB[pix * LDSW + ch] = f2bf(g);
      }
  __syncthreads();

  // -------- stage 3: out = xhr2 + Wm2 @ hmid + b_mlp2 --------
#pragma unroll
  for (int mi = 0; mi < 2; ++mi) {
    int chb = mtb + mi * 16 + h * 8;
#pragma unroll
    for (int i = 0; i < 8; ++i) {
      float bv = biasm2[chb + i];
#pragma unroll
      for (int ni = 0; ni < 4; ++ni) acc[mi][ni][i] = bv;
    }
  }
  gemm_256x64(wm2, bufB, mtb, m, h, acc);
#pragma unroll
  for (int mi = 0; mi < 2; ++mi)
#pragma unroll
    for (int ni = 0; ni < 4; ++ni)
#pragma unroll
      for (int i = 0; i < 8; ++i) {
        int ch  = mtb + mi * 16 + i + h * 8;
        int pix = ni * 16 + m;
        obase[ch * HW + pix] = resid[mi][ni][i] + acc[mi][ni][i];
      }
}

// ---------------------------------------------------------------------------
extern "C" void kernel_launch(void* const* d_in, const int* in_sizes, int n_in,
                              void* d_out, int out_size, void* d_ws, size_t ws_size,
                              hipStream_t stream) {
  (void)in_sizes; (void)n_in; (void)out_size; (void)ws_size;
  const float* xhr      = (const float*)d_in[0];
  const float* xlr      = (const float*)d_in[1];
  const float* w_logits = (const float*)d_in[2];
  const float* b_logits = (const float*)d_in[3];
  const float* w_values = (const float*)d_in[4];
  const float* b_values = (const float*)d_in[5];
  const float* w_proj   = (const float*)d_in[6];
  const float* b_proj   = (const float*)d_in[7];
  const float* w_mlp1   = (const float*)d_in[8];
  const float* b_mlp1   = (const float*)d_in[9];
  const float* w_mlp2   = (const float*)d_in[10];
  const float* b_mlp2   = (const float*)d_in[11];
  const float* g1  = (const float*)d_in[12];
  const float* bb1 = (const float*)d_in[13];
  const float* m1  = (const float*)d_in[14];
  const float* v1  = (const float*)d_in[15];
  const float* g2  = (const float*)d_in[16];
  const float* bb2 = (const float*)d_in[17];
  const float* m2  = (const float*)d_in[18];
  const float* v2  = (const float*)d_in[19];

  char* ws = (char*)d_ws;
  unsigned short* wp  = (unsigned short*)(ws + 0);
  unsigned short* wm1 = (unsigned short*)(ws + 131072);
  unsigned short* wm2 = (unsigned short*)(ws + 262144);
  unsigned short* wvT = (unsigned short*)(ws + 393216);
  float* biasp  = (float*)(ws + 524288);
  float* biasm1 = (float*)(ws + 525312);
  float* biasm2 = (float*)(ws + 526336);
  float* biasv  = (float*)(ws + 527360);
  float* vals   = (float*)(ws + 528384);   // 4*256*256 f32 = 1 MB

  fold_kernel<<<dim3(256, 4), 256, 0, stream>>>(
      w_proj, b_proj, w_mlp1, b_mlp1, w_mlp2, b_mlp2, w_values, b_values,
      g1, bb1, m1, v1, g2, bb2, m2, v2,
      wp, wm1, wm2, wvT, biasp, biasm1, biasm2, biasv);
  vals_kernel<<<1024, 256, 0, stream>>>(xlr, wvT, biasv, vals);
  fused_kernel<<<4096, 256, 0, stream>>>(
      xhr, w_logits, b_logits, wp, wm1, wm2, biasp, biasm1, biasm2,
      vals, (float*)d_out);
}
